// GraphSAGE_35330400977261
// MI455X (gfx1250) — compile-verified, matrix operand order checked by reading, not run on hardware
//
#include <hip/hip_runtime.h>

typedef __attribute__((ext_vector_type(16))) _Float16 v16h;
typedef __attribute__((ext_vector_type(8)))  float    v8f;

#define NN 50000
#define EE 800000
#define KD 128

// ---------------- utility kernels ----------------

__global__ void k_zero(float* __restrict__ p, int n) {
  int i = blockIdx.x * blockDim.x + threadIdx.x;
  int s = gridDim.x * blockDim.x;
  for (; i < n; i += s) p[i] = 0.0f;
}

__global__ void k_deg(const int* __restrict__ dst, float* __restrict__ deg, int e) {
  int i = blockIdx.x * blockDim.x + threadIdx.x;
  if (i < e) unsafeAtomicAdd(&deg[dst[i]], 1.0f);
}

__global__ void k_deginv(float* __restrict__ deg, int n) {
  int i = blockIdx.x * blockDim.x + threadIdx.x;
  if (i < n) deg[i] = 1.0f / fmaxf(deg[i], 1.0f);
}

// Convert f32 weight [K, Nout] into WMMA-B fragment-major f16 layout:
//   out[((tcol*4 + kb)*32 + lane)*16 + j]
// so each lane's 16 halves for one 16x16x32 WMMA step are contiguous (32B).
__global__ void k_swizzleW(const float* __restrict__ W, _Float16* __restrict__ out,
                           int Nout) {
  int idx = blockIdx.x * blockDim.x + threadIdx.x;   // KD * Nout elements
  if (idx >= KD * Nout) return;
  int j    = idx & 15;
  int lane = (idx >> 4) & 31;
  int kb   = (idx >> 9) & 3;
  int tcol = idx >> 11;
  int v    = j >> 1;
  int e    = j & 1;
  int half = lane >> 4;
  int n    = tcol * 16 + (lane & 15);
  int k    = kb * 32 + ((v & 4) << 2) + (half << 3) + ((v & 3) << 1) + e;
  out[idx] = (_Float16)W[k * Nout + n];
}

// scatter-add aggregation: one lane handles 4 consecutive floats of one edge row
__global__ void k_agg(const float* __restrict__ h, const int* __restrict__ src,
                      const int* __restrict__ dst, float* __restrict__ msum, int e) {
  int idx = blockIdx.x * blockDim.x + threadIdx.x;   // e * 32 threads
  int eid = idx >> 5;
  if (eid >= e) return;
  int f = (idx & 31) << 2;
  int s = src[eid];
  int d = dst[eid];
  const float4 v = *(const float4*)(h + (long long)s * KD + f);
  float* o = msum + (long long)d * KD + f;
  unsafeAtomicAdd(o + 0, v.x);
  unsafeAtomicAdd(o + 1, v.y);
  unsafeAtomicAdd(o + 2, v.z);
  unsafeAtomicAdd(o + 3, v.w);
}

__device__ __forceinline__ void cvt8(v16h& dst, int base, float4 x, float4 y, float s) {
  dst[base + 0] = (_Float16)(x.x * s);
  dst[base + 1] = (_Float16)(x.y * s);
  dst[base + 2] = (_Float16)(x.z * s);
  dst[base + 3] = (_Float16)(x.w * s);
  dst[base + 4] = (_Float16)(y.x * s);
  dst[base + 5] = (_Float16)(y.y * s);
  dst[base + 6] = (_Float16)(y.z * s);
  dst[base + 7] = (_Float16)(y.w * s);
}

// ---------------- fused dual-GEMM SAGE layer ----------------
// out[row, coff+col] = act( Aself@Ws + (msum*deg_inv)@Wn + bias )
// optional out2 = out + noise  (second stream for the noise path)
__global__ __launch_bounds__(256)
void k_sage_gemm(const float* __restrict__ Aself,
                 const float* __restrict__ msum,
                 const float* __restrict__ deg_inv,
                 const _Float16* __restrict__ WsF,   // fragment-major
                 const _Float16* __restrict__ WnF,   // fragment-major
                 const float* __restrict__ bias,
                 const float* __restrict__ noise,    // may be null
                 float* __restrict__ out, int out_ld, int out_coff,
                 float* __restrict__ out2,           // may be null
                 int M, int Nout, int act) {
  const int ctiles = Nout >> 4;
  const int total  = (M >> 4) * ctiles;
  const int wave   = (blockIdx.x << 3) + (threadIdx.x >> 5);
  if (wave >= total) return;

  const int trow = wave / ctiles;
  const int tcol = wave - trow * ctiles;
  const int lane = threadIdx.x & 31;
  const int half = lane >> 4;
  const int mloc = lane & 15;
  const int row  = trow * 16 + mloc;
  const int ncol = tcol * 16 + mloc;

  const float dinv = deg_inv[row];
  const float* __restrict__ arow = Aself + (long long)row * KD;
  const float* __restrict__ grow = msum  + (long long)row * KD;
  // per-lane contiguous B fragments
  const _Float16* __restrict__ bsp = WsF + (((tcol * 4) * 32 + lane) << 4);
  const _Float16* __restrict__ bnp = WnF + (((tcol * 4) * 32 + lane) << 4);

  v8f acc_s = {};
  v8f acc_n = {};
#pragma unroll
  for (int kb = 0; kb < 4; ++kb) {
    if (kb < 3) {
      __builtin_prefetch(arow + (kb + 1) * 32, 0, 3);
      __builtin_prefetch(grow + (kb + 1) * 32, 0, 3);
    }
    const int kbase = kb * 32 + (half << 3);
    const float4 a0 = *(const float4*)(arow + kbase);
    const float4 a1 = *(const float4*)(arow + kbase + 4);
    const float4 a2 = *(const float4*)(arow + kbase + 16);
    const float4 a3 = *(const float4*)(arow + kbase + 20);
    const float4 g0 = *(const float4*)(grow + kbase);
    const float4 g1 = *(const float4*)(grow + kbase + 4);
    const float4 g2 = *(const float4*)(grow + kbase + 16);
    const float4 g3 = *(const float4*)(grow + kbase + 20);

    v16h a, an;
    cvt8(a,  0, a0, a1, 1.0f);
    cvt8(a,  8, a2, a3, 1.0f);
    cvt8(an, 0, g0, g1, dinv);
    cvt8(an, 8, g2, g3, dinv);

    const v16h bs = *(const v16h*)(bsp + (kb << 9));   // kb*32 frags * 16 halves
    const v16h bn = *(const v16h*)(bnp + (kb << 9));

    // independent accumulators -> no back-to-back WMMA RAW chain
    acc_s = __builtin_amdgcn_wmma_f32_16x16x32_f16(false, a,  false, bs, (short)0, acc_s, false, false);
    acc_n = __builtin_amdgcn_wmma_f32_16x16x32_f16(false, an, false, bn, (short)0, acc_n, false, false);
  }

  const float bv = bias[ncol];
#pragma unroll
  for (int r = 0; r < 8; ++r) {
    const int m  = r + (half << 3);                    // C/D: VGPR r -> row r + 8*half
    const int gr = trow * 16 + m;
    float val = acc_s[r] + acc_n[r] + bv;
    if (act) val = fmaxf(val, 0.0f);
    out[(long long)gr * out_ld + out_coff + ncol] = val;
    if (out2)
      out2[(long long)gr * out_ld + out_coff + ncol] =
          val + noise[(long long)gr * KD + ncol];
  }
}

// ---------------- host orchestration ----------------

extern "C" void kernel_launch(void* const* d_in, const int* in_sizes, int n_in,
                              void* d_out, int out_size, void* d_ws, size_t ws_size,
                              hipStream_t stream) {
  const float* features = (const float*)d_in[0];
  const float* noise    = (const float*)d_in[1];
  const float* Ws0 = (const float*)d_in[2];
  const float* Wn0 = (const float*)d_in[3];
  const float* b0  = (const float*)d_in[4];
  const float* Ws1 = (const float*)d_in[5];
  const float* Wn1 = (const float*)d_in[6];
  const float* b1  = (const float*)d_in[7];
  const float* Ws2 = (const float*)d_in[8];
  const float* Wn2 = (const float*)d_in[9];
  const float* b2  = (const float*)d_in[10];
  const int* esrc  = (const int*)d_in[11];
  const int* edst  = (const int*)d_in[12];
  float* out = (float*)d_out;

  float* ws   = (float*)d_ws;
  float* deg  = ws;                       // NN (becomes deg_inv in place)
  float* msum = ws + 50048;               // NN*KD, reused for every aggregation
  float* h0   = msum + NN * KD;
  float* h1   = h0   + NN * KD;
  float* hn1  = h1   + NN * KD;
  _Float16* Ws0h = (_Float16*)(hn1 + NN * KD);
  _Float16* Wn0h = Ws0h + KD * 128;
  _Float16* Ws1h = Wn0h + KD * 128;
  _Float16* Wn1h = Ws1h + KD * 128;
  _Float16* Ws2h = Wn1h + KD * 128;
  _Float16* Wn2h = Ws2h + KD * 64;

  // degrees -> deg_inv
  k_zero<<<256, 256, 0, stream>>>(deg, NN);
  k_deg<<<(EE + 255) / 256, 256, 0, stream>>>(edst, deg, EE);
  k_deginv<<<(NN + 255) / 256, 256, 0, stream>>>(deg, NN);

  // fragment-major f16 weight copies (tiny, once per call)
  k_swizzleW<<<64, 256, 0, stream>>>(Ws0, Ws0h, 128);
  k_swizzleW<<<64, 256, 0, stream>>>(Wn0, Wn0h, 128);
  k_swizzleW<<<64, 256, 0, stream>>>(Ws1, Ws1h, 128);
  k_swizzleW<<<64, 256, 0, stream>>>(Wn1, Wn1h, 128);
  k_swizzleW<<<32, 256, 0, stream>>>(Ws2, Ws2h, 64);
  k_swizzleW<<<32, 256, 0, stream>>>(Wn2, Wn2h, 64);

  const int aggBlocks  = (EE * 32 + 255) / 256;
  const int gemm128Blk = (NN / 16) * (128 / 16) / 8;          // 3125
  const int gemm64Blk  = ((NN / 16) * (64 / 16) + 7) / 8;     // 1563

  // layer 0: h0 = relu(features@Ws0 + mean@Wn0 + b0)
  k_zero<<<2048, 256, 0, stream>>>(msum, NN * KD);
  k_agg<<<aggBlocks, 256, 0, stream>>>(features, esrc, edst, msum, EE);
  k_sage_gemm<<<gemm128Blk, 256, 0, stream>>>(features, msum, deg, Ws0h, Wn0h, b0,
                                              nullptr, h0, 128, 0, nullptr, NN, 128, 1);

  // layer 1: h1 = relu(...), hn1 = h1 + noise  (noise_d == 1)
  k_zero<<<2048, 256, 0, stream>>>(msum, NN * KD);
  k_agg<<<aggBlocks, 256, 0, stream>>>(h0, esrc, edst, msum, EE);
  k_sage_gemm<<<gemm128Blk, 256, 0, stream>>>(h0, msum, deg, Ws1h, Wn1h, b1,
                                              noise, h1, 128, 0, hn1, NN, 128, 1);

  // layer 2 clean path -> out[:, 64:128]
  k_zero<<<2048, 256, 0, stream>>>(msum, NN * KD);
  k_agg<<<aggBlocks, 256, 0, stream>>>(h1, esrc, edst, msum, EE);
  k_sage_gemm<<<gemm64Blk, 256, 0, stream>>>(h1, msum, deg, Ws2h, Wn2h, b2,
                                             nullptr, out, 128, 64, nullptr, NN, 64, 0);

  // layer 2 noise path -> out[:, 0:64]
  k_zero<<<2048, 256, 0, stream>>>(msum, NN * KD);
  k_agg<<<aggBlocks, 256, 0, stream>>>(hn1, esrc, edst, msum, EE);
  k_sage_gemm<<<gemm64Blk, 256, 0, stream>>>(hn1, msum, deg, Ws2h, Wn2h, b2,
                                             nullptr, out, 128, 0, nullptr, NN, 64, 0);
}
